// RSA_45853070852588
// MI455X (gfx1250) — compile-verified, matrix operand order checked by reading, not run on hardware
//
#include <hip/hip_runtime.h>
#include <stdint.h>

typedef __attribute__((ext_vector_type(16))) __bf16 v16bf;
typedef __attribute__((ext_vector_type(8)))  float  v8f;

union Frag {
  v16bf v;
  uint4 q[2];
};

__device__ __forceinline__ unsigned short f2bf(float f) {
  union { float f; uint32_t u; } x; x.f = f;
  uint32_t r = x.u + 0x7fffu + ((x.u >> 16) & 1u);
  return (unsigned short)(r >> 16);
}

// Async 16-byte copy global -> LDS (CDNA5 GLOBAL_LOAD_ASYNC_TO_LDS_B128,
// tracked by ASYNCcnt). LDS byte address = low 32 bits of the generic pointer
// (ISA 10.2: LDS aperture truncates to addr[31:0]).
__device__ __forceinline__ void asy16(void* lds_dst, const void* gsrc) {
  unsigned loff = (unsigned)(size_t)lds_dst;
  asm volatile("global_load_async_to_lds_b128 %0, %1, off"
               :: "v"(loff), "v"(gsrc)
               : "memory");
}

__device__ __forceinline__ void wait_async0() {
  asm volatile("s_wait_asynccnt 0x0" ::: "memory");
}

// ---------------------------------------------------------------------------
// Big GEMM: C[M,N] = A[M,K](bf16) * Bt[N,K](bf16)^T (+bias)
// 64x64 block tile, 4 waves (2x2), each wave 32x32 via 4 WMMA accumulators.
// Double-buffered LDS fed by async-to-LDS copies; one barrier per K-step,
// copy of tile t+1 overlaps WMMA of tile t.
// mode 0: f32 row-major (+bias). mode 1: bf16 row-major. mode 2: bf16
// per-batch transposed store: out[(m/tposRows)*N + n][m%tposRows].
// Requires M%64==0, N%64==0, K%32==0.
// ---------------------------------------------------------------------------
__global__ __launch_bounds__(128) void k_gemm64(
    const unsigned short* __restrict__ A, int lda,
    const unsigned short* __restrict__ Bt, int ldb,
    const float* __restrict__ bias,
    void* __restrict__ C, int ldc,
    int M, int N, int K, int mode, int tposRows)
{
  (void)M;
  __shared__ __align__(16) unsigned short sA[2][64 * 32];
  __shared__ __align__(16) unsigned short sB[2][64 * 32];
  const int tid  = threadIdx.x;
  const int lane = tid & 31;
  const int wave = tid >> 5;
  const int wm = (wave & 1) * 32;
  const int wn = (wave >> 1) * 32;
  const size_t bm = (size_t)blockIdx.x * 64;
  const size_t bn = (size_t)blockIdx.y * 64;

  // per-thread staging coordinates: thread covers rows r0 and r0+32, 8 cols
  const int r0 = tid >> 2;
  const int c0 = (tid & 3) * 8;
  const unsigned short* gA0 = A + (bm + r0) * (size_t)lda + c0;
  const unsigned short* gA1 = A + (bm + r0 + 32) * (size_t)lda + c0;
  const unsigned short* gB0 = Bt + (bn + r0) * (size_t)ldb + c0;
  const unsigned short* gB1 = Bt + (bn + r0 + 32) * (size_t)ldb + c0;

  auto issueTile = [&](int buf, int k0) {
    asy16(&sA[buf][r0 * 32 + c0],        gA0 + k0);
    asy16(&sA[buf][(r0 + 32) * 32 + c0], gA1 + k0);
    asy16(&sB[buf][r0 * 32 + c0],        gB0 + k0);
    asy16(&sB[buf][(r0 + 32) * 32 + c0], gB1 + k0);
  };

  v8f acc[2][2] = {};
  const int T = K >> 5;
  issueTile(0, 0);

  for (int t = 0; t < T; ++t) {
    wait_async0();        // own async writes for tile t are in LDS
    __syncthreads();      // everyone's writes visible; prior reads of buf^1 done
    const int buf = t & 1;
    if (t + 1 < T) issueTile(buf ^ 1, (t + 1) << 5);

    const int half = lane >> 4;
    const int l15  = lane & 15;
    Frag a[2], b[2];
#pragma unroll
    for (int i = 0; i < 2; ++i) {
      const unsigned short* pa = &sA[buf][(wm + i * 16 + l15) * 32 + half * 8];
      a[i].q[0] = *(const uint4*)pa;
      a[i].q[1] = *(const uint4*)(pa + 16);
      const unsigned short* pb = &sB[buf][(wn + i * 16 + l15) * 32 + half * 8];
      b[i].q[0] = *(const uint4*)pb;
      b[i].q[1] = *(const uint4*)(pb + 16);
    }
#pragma unroll
    for (int i = 0; i < 2; ++i)
#pragma unroll
      for (int j = 0; j < 2; ++j)
        acc[i][j] = __builtin_amdgcn_wmma_f32_16x16x32_bf16(
            false, a[i].v, false, b[j].v, (short)0, acc[i][j], false, false);
  }

  const int half = lane >> 4;
  const int l15  = lane & 15;
#pragma unroll
  for (int i = 0; i < 2; ++i) {
#pragma unroll
    for (int j = 0; j < 2; ++j) {
      size_t n = bn + wn + j * 16 + l15;
      float bv = bias ? bias[n] : 0.0f;
#pragma unroll
      for (int r = 0; r < 8; ++r) {
        size_t m = bm + wm + i * 16 + half * 8 + r;
        float val = acc[i][j][r] + bv;
        if (mode == 0) {
          ((float*)C)[m * (size_t)ldc + n] = val;
        } else if (mode == 1) {
          ((unsigned short*)C)[m * (size_t)ldc + n] = f2bf(val);
        } else {
          size_t bb = m / (size_t)tposRows;
          size_t jj = m % (size_t)tposRows;
          ((unsigned short*)C)[(bb * (size_t)N + n) * (size_t)tposRows + jj] = f2bf(val);
        }
      }
    }
  }
}

// ---------------------------------------------------------------------------
// Skinny batched GEMM: per batch z, C[16,N] = A[16,K] * Bt[N,K]^T.
// 16x128 block tile, 4 waves, each wave 16x32. Double-buffered async-to-LDS.
// Stores rows m < Mstore.
// ---------------------------------------------------------------------------
__global__ __launch_bounds__(128) void k_gemm_skinny(
    const unsigned short* __restrict__ A, size_t sAb, int lda,
    const unsigned short* __restrict__ Bt, size_t sBb, int ldb,
    float* __restrict__ C, size_t sCb, int ldc,
    int N, int K, int Mstore)
{
  (void)N;
  __shared__ __align__(16) unsigned short shA[2][16 * 32];
  __shared__ __align__(16) unsigned short shB[2][128 * 32];
  const int bz = blockIdx.z;
  A  += (size_t)bz * sAb;
  Bt += (size_t)bz * sBb;
  C  += (size_t)bz * sCb;
  const size_t bn = (size_t)blockIdx.x * 128;
  const int tid = threadIdx.x, lane = tid & 31, wave = tid >> 5;
  const int wn = wave * 32;

  const int r0 = tid >> 2;            // 0..31
  const int c0 = (tid & 3) * 8;
  const unsigned short* gA = A + (size_t)r0 * lda + c0;          // valid if r0<16
  const unsigned short* gB0 = Bt + (bn + r0) * (size_t)ldb + c0;

  auto issueTile = [&](int buf, int k0) {
    if (tid < 64) asy16(&shA[buf][r0 * 32 + c0], gA + k0);
#pragma unroll
    for (int it = 0; it < 4; ++it) {
      int r = r0 + it * 32;
      asy16(&shB[buf][r * 32 + c0], gB0 + (size_t)it * 32 * ldb + k0);
    }
  };

  v8f acc[2] = {};
  const int T = K >> 5;
  issueTile(0, 0);

  for (int t = 0; t < T; ++t) {
    wait_async0();
    __syncthreads();
    const int buf = t & 1;
    if (t + 1 < T) issueTile(buf ^ 1, (t + 1) << 5);

    const int half = lane >> 4, l15 = lane & 15;
    Frag a, b[2];
    const unsigned short* pa = &shA[buf][l15 * 32 + half * 8];
    a.q[0] = *(const uint4*)pa;
    a.q[1] = *(const uint4*)(pa + 16);
#pragma unroll
    for (int j = 0; j < 2; ++j) {
      const unsigned short* pb = &shB[buf][(wn + j * 16 + l15) * 32 + half * 8];
      b[j].q[0] = *(const uint4*)pb;
      b[j].q[1] = *(const uint4*)(pb + 16);
      acc[j] = __builtin_amdgcn_wmma_f32_16x16x32_bf16(
          false, a.v, false, b[j].v, (short)0, acc[j], false, false);
    }
  }
  const int half = lane >> 4, l15 = lane & 15;
#pragma unroll
  for (int j = 0; j < 2; ++j) {
    size_t n = bn + wn + j * 16 + l15;
#pragma unroll
    for (int r = 0; r < 8; ++r) {
      int m = half * 8 + r;
      if (m < Mstore) C[(size_t)m * ldc + n] = acc[j][r];
    }
  }
}

// ---------------------------------------------------------------------------
// Row-wise LayerNorm: one 256-thread block per row (wave32 shfl reduce).
// Optional f32 output and/or bf16 output.
// ---------------------------------------------------------------------------
__global__ __launch_bounds__(256) void k_ln_rows(
    const float* __restrict__ x, const float* __restrict__ g, const float* __restrict__ b,
    float* __restrict__ outF, unsigned short* __restrict__ outB, int d)
{
  const size_t row = blockIdx.x;
  const float* xr = x + row * (size_t)d;
  const int tid = threadIdx.x, lane = tid & 31, wave = tid >> 5;
  __shared__ float rs[8], rs2[8];
  __shared__ float sMu, sRstd;
  float s = 0.f, s2 = 0.f;
  for (int c = tid; c < d; c += 256) { float v = xr[c]; s += v; s2 += v * v; }
  for (int o = 16; o > 0; o >>= 1) { s += __shfl_xor(s, o); s2 += __shfl_xor(s2, o); }
  if (lane == 0) { rs[wave] = s; rs2[wave] = s2; }
  __syncthreads();
  if (tid == 0) {
    float ts = 0.f, ts2 = 0.f;
    for (int w = 0; w < 8; ++w) { ts += rs[w]; ts2 += rs2[w]; }
    float mu = ts / d;
    float var = ts2 / d - mu * mu;
    sMu = mu;
    sRstd = rsqrtf(var + 1e-5f);
  }
  __syncthreads();
  const float mu = sMu, rstd = sRstd;
  for (int c = tid; c < d; c += 256) {
    float y = (xr[c] - mu) * rstd * g[c] + b[c];
    if (outF) outF[row * (size_t)d + c] = y;
    if (outB) outB[row * (size_t)d + c] = f2bf(y);
  }
}

// --------------------------- small elementwise ------------------------------

__global__ void k_cvt_bf16(const float* __restrict__ s, unsigned short* __restrict__ d, int n) {
  int i = blockIdx.x * 256 + threadIdx.x;
  if (i < n) d[i] = f2bf(s[i]);
}

__global__ void k_transpose_bf16(const float* __restrict__ src, unsigned short* __restrict__ dst,
                                 int R, int C, int Rp, int Cp) {
  int i = blockIdx.x * 256 + threadIdx.x;
  if (i >= Rp * Cp) return;
  int c = i / Rp, r = i % Rp;
  float v = (r < R && c < C) ? src[(size_t)r * C + c] : 0.0f;
  dst[(size_t)c * Rp + r] = f2bf(v);
}

__global__ void k_pad_bias(const float* __restrict__ s, float* __restrict__ d, int n, int np) {
  int i = blockIdx.x * 256 + threadIdx.x;
  if (i < np) d[i] = (i < n) ? s[i] : 0.0f;
}

__global__ void k_init_slots(const float* __restrict__ noise, const float* __restrict__ mu,
                             const float* __restrict__ ls, float* __restrict__ out, int n, int d) {
  int i = blockIdx.x * 256 + threadIdx.x;
  if (i >= n) return;
  int c = i % d;
  out[i] = mu[c] + expf(ls[c]) * noise[i];
}

__global__ void k_pack_q(const float* __restrict__ q, unsigned short* __restrict__ qp, int total) {
  int i = blockIdx.x * 256 + threadIdx.x;
  if (i >= total) return;
  int row = i >> 9, col = i & 511;
  int bb = row >> 3, sidx = row & 7;
  qp[(((size_t)bb * 16 + sidx) << 9) + col] = f2bf(q[i]);
}

// softmax over the 8 slots per (batch, position); * scale; + eps
__global__ void k_softmax_col(float* __restrict__ dots) {
  const float SCALE_QK = 0.03608439182435161f; // 768^-0.5
  int t = blockIdx.x * 256 + threadIdx.x;
  if (t >= 32 * 4096) return;
  int bb = t >> 12, j = t & 4095;
  float* p = dots + ((size_t)bb * 16) * 4096 + j;
  float v[8], mx = -1e30f;
#pragma unroll
  for (int i = 0; i < 8; ++i) { v[i] = p[(size_t)i * 4096] * SCALE_QK; mx = fmaxf(mx, v[i]); }
  float s = 0.f;
#pragma unroll
  for (int i = 0; i < 8; ++i) { v[i] = expf(v[i] - mx); s += v[i]; }
  float inv = 1.0f / s;
#pragma unroll
  for (int i = 0; i < 8; ++i) p[(size_t)i * 4096] = v[i] * inv + 1e-8f;
}

__global__ __launch_bounds__(256) void k_rowsum(const float* __restrict__ dots, float* __restrict__ sums) {
  int bi = blockIdx.x;               // b*8 + i
  int bb = bi >> 3, i = bi & 7;
  const float* p = dots + ((size_t)bb * 16 + i) * 4096;
  int tid = threadIdx.x, lane = tid & 31, wave = tid >> 5;
  __shared__ float rs[8];
  float s = 0.f;
  for (int j = tid; j < 4096; j += 256) s += p[j];
  for (int o = 16; o > 0; o >>= 1) s += __shfl_xor(s, o);
  if (lane == 0) rs[wave] = s;
  __syncthreads();
  if (tid == 0) {
    float t = 0.f;
    for (int w = 0; w < 8; ++w) t += rs[w];
    sums[bi] = t;
  }
}

__global__ void k_attn_norm(const float* __restrict__ dots, const float* __restrict__ sums,
                            unsigned short* __restrict__ attn, float* __restrict__ outAttn) {
  int t = blockIdx.x * 256 + threadIdx.x;
  if (t >= 32 * 8 * 4096) return;
  int j = t & 4095;
  int i = (t >> 12) & 7;
  int bb = t >> 15;
  float v = dots[((size_t)bb * 16 + i) * 4096 + j] / sums[bb * 8 + i];
  attn[((size_t)bb * 16 + i) * 4096 + j] = f2bf(v);
  if (outAttn) outAttn[((size_t)bb * 4096 + j) * 8 + i] = v;
}

__global__ void k_gru_combine(const float* __restrict__ gi, const float* __restrict__ gh,
                              const float* __restrict__ h, float* __restrict__ out, int n) {
  int t = blockIdx.x * 256 + threadIdx.x;
  if (t >= n) return;
  int row = t >> 9, col = t & 511;
  size_t base = (size_t)row * 1536;
  float r = 1.f / (1.f + expf(-(gi[base + col] + gh[base + col])));
  float z = 1.f / (1.f + expf(-(gi[base + 512 + col] + gh[base + 512 + col])));
  float nn = tanhf(gi[base + 1024 + col] + r * gh[base + 1024 + col]);
  out[t] = (1.f - z) * nn + z * h[t];
}

__global__ void k_swiglu(const float* __restrict__ h1, const float* __restrict__ h2,
                         unsigned short* __restrict__ g, int n) {
  int t = blockIdx.x * 256 + threadIdx.x;
  if (t >= n) return;
  float a = h1[t];
  float si = a / (1.f + expf(-a));
  g[t] = f2bf(si * h2[t]);
}

__global__ void k_add(float* __restrict__ dst, const float* __restrict__ src, int n) {
  int t = blockIdx.x * 256 + threadIdx.x;
  if (t < n) dst[t] += src[t];
}

__global__ void k_copyf(float* __restrict__ dst, const float* __restrict__ src, int n) {
  int t = blockIdx.x * 256 + threadIdx.x;
  if (t < n) dst[t] = src[t];
}

// ---------------------------------------------------------------------------

extern "C" void kernel_launch(void* const* d_in, const int* in_sizes, int n_in,
                              void* d_out, int out_size, void* d_ws, size_t ws_size,
                              hipStream_t stream)
{
  (void)in_sizes; (void)n_in; (void)out_size; (void)ws_size;
  const float* x       = (const float*)d_in[0];
  const float* noise   = (const float*)d_in[1];
  const float* ln_in_g = (const float*)d_in[2];
  const float* ln_in_b = (const float*)d_in[3];
  const float* init_mu = (const float*)d_in[4];
  const float* init_ls = (const float*)d_in[5];
  const float* Wk      = (const float*)d_in[6];
  const float* Wv      = (const float*)d_in[7];
  const float* Wq      = (const float*)d_in[8];
  const float* gWih    = (const float*)d_in[9];
  const float* gWhh    = (const float*)d_in[10];
  const float* gbih    = (const float*)d_in[11];
  const float* gbhh    = (const float*)d_in[12];
  const float* ns_g    = (const float*)d_in[13];
  const float* ns_b    = (const float*)d_in[14];
  const float* ni_g    = (const float*)d_in[15];
  const float* ni_b    = (const float*)d_in[16];
  const float* f_w1    = (const float*)d_in[17];
  const float* f_b1    = (const float*)d_in[18];
  const float* f_w2    = (const float*)d_in[19];
  const float* f_b2    = (const float*)d_in[20];
  const float* f_w3    = (const float*)d_in[21];
  const float* f_b3    = (const float*)d_in[22];

  const int B = 32, NN = 4096, DIN = 768, D = 512, NS = 8, H = 1365, HP = 1408, RR = 256, G3 = 1536;
  const int MROW = B * NN; // 131072

  char* base = (char*)d_ws;
  size_t off = 0;
  auto alloc = [&](size_t bytes) -> char* {
    char* p = base + off;
    off = (off + bytes + 255) & ~(size_t)255;
    return p;
  };
  unsigned short* xln    = (unsigned short*)alloc((size_t)MROW * DIN * 2);
  unsigned short* kbuf   = (unsigned short*)alloc((size_t)MROW * D * 2);
  unsigned short* vt     = (unsigned short*)alloc((size_t)MROW * D * 2);
  unsigned short* Wkt    = (unsigned short*)alloc((size_t)D * DIN * 2);
  unsigned short* Wvt    = (unsigned short*)alloc((size_t)D * DIN * 2);
  unsigned short* Wqt    = (unsigned short*)alloc((size_t)D * D * 2);
  unsigned short* Wihb   = (unsigned short*)alloc((size_t)G3 * D * 2);
  unsigned short* Whhb   = (unsigned short*)alloc((size_t)G3 * D * 2);
  unsigned short* w1t    = (unsigned short*)alloc((size_t)4 * HP * D * 2);
  unsigned short* w2t    = (unsigned short*)alloc((size_t)4 * HP * D * 2);
  unsigned short* w3t    = (unsigned short*)alloc((size_t)4 * D * HP * 2);
  float* b1p             = (float*)alloc((size_t)4 * HP * 4);
  float* b2p             = (float*)alloc((size_t)4 * HP * 4);
  float* slotsF          = (float*)alloc((size_t)RR * D * 4);
  unsigned short* slotsB = (unsigned short*)alloc((size_t)RR * D * 2);
  unsigned short* qln    = (unsigned short*)alloc((size_t)RR * D * 2);
  float* qF              = (float*)alloc((size_t)RR * D * 4);
  unsigned short* qpad   = (unsigned short*)alloc((size_t)B * 16 * D * 2);
  float* dots            = (float*)alloc((size_t)B * 16 * NN * 4);
  float* sums            = (float*)alloc((size_t)RR * 4);
  unsigned short* attnB  = (unsigned short*)alloc((size_t)B * 16 * NN * 2);
  float* updF            = (float*)alloc((size_t)RR * D * 4);
  unsigned short* updB   = (unsigned short*)alloc((size_t)RR * D * 2);
  float* gi              = (float*)alloc((size_t)RR * G3 * 4);
  float* gh              = (float*)alloc((size_t)RR * G3 * 4);
  float* newh            = (float*)alloc((size_t)RR * D * 4);
  float* h1              = (float*)alloc((size_t)RR * HP * 4);
  float* h2              = (float*)alloc((size_t)RR * HP * 4);
  unsigned short* gact   = (unsigned short*)alloc((size_t)RR * HP * 2);
  float* ffno            = (float*)alloc((size_t)RR * D * 4);

  auto grid1 = [](int n) { return dim3((unsigned)((n + 255) / 256)); };

  // zero padded slot-row buffers (rows 8..15 must be zero for the M=16 GEMMs)
  hipMemsetAsync(qpad, 0, (size_t)B * 16 * D * 2, stream);
  hipMemsetAsync(attnB, 0, (size_t)B * 16 * NN * 2, stream);

  // ---- weight prep (f32 -> bf16, transposed into Bt[N][K] layout) ----
  k_transpose_bf16<<<grid1(DIN * D), 256, 0, stream>>>(Wk, Wkt, DIN, D, DIN, D);
  k_transpose_bf16<<<grid1(DIN * D), 256, 0, stream>>>(Wv, Wvt, DIN, D, DIN, D);
  k_transpose_bf16<<<grid1(D * D), 256, 0, stream>>>(Wq, Wqt, D, D, D, D);
  k_cvt_bf16<<<grid1(G3 * D), 256, 0, stream>>>(gWih, Wihb, G3 * D);
  k_cvt_bf16<<<grid1(G3 * D), 256, 0, stream>>>(gWhh, Whhb, G3 * D);
  for (int l = 0; l < 4; ++l) {
    k_transpose_bf16<<<grid1(HP * D), 256, 0, stream>>>(f_w1 + (size_t)l * D * H, w1t + (size_t)l * HP * D, D, H, D, HP);
    k_transpose_bf16<<<grid1(HP * D), 256, 0, stream>>>(f_w2 + (size_t)l * D * H, w2t + (size_t)l * HP * D, D, H, D, HP);
    k_transpose_bf16<<<grid1(HP * D), 256, 0, stream>>>(f_w3 + (size_t)l * H * D, w3t + (size_t)l * D * HP, H, D, HP, D);
    k_pad_bias<<<grid1(HP), 256, 0, stream>>>(f_b1 + (size_t)l * H, b1p + (size_t)l * HP, H, HP);
    k_pad_bias<<<grid1(HP), 256, 0, stream>>>(f_b2 + (size_t)l * H, b2p + (size_t)l * HP, H, HP);
  }

  // ---- input LN -> bf16, then k and v projections (v stored transposed) ----
  k_ln_rows<<<MROW, 256, 0, stream>>>(x, ln_in_g, ln_in_b, nullptr, xln, DIN);
  k_gemm64<<<dim3(MROW / 64, D / 64), 128, 0, stream>>>(
      xln, DIN, Wkt, DIN, nullptr, kbuf, D, MROW, D, DIN, 1, 0);
  k_gemm64<<<dim3(MROW / 64, D / 64), 128, 0, stream>>>(
      xln, DIN, Wvt, DIN, nullptr, vt, 0, MROW, D, DIN, 2, NN);

  // ---- init slots ----
  k_init_slots<<<grid1(RR * D), 256, 0, stream>>>(noise, init_mu, init_ls, slotsF, RR * D, D);

  for (int l = 0; l < 4; ++l) {
    // slots bf16 (GRU hidden path) + q = LN_ns(slots) @ Wq
    k_cvt_bf16<<<grid1(RR * D), 256, 0, stream>>>(slotsF, slotsB, RR * D);
    k_ln_rows<<<RR, 256, 0, stream>>>(slotsF, ns_g, ns_b, nullptr, qln, D);
    k_gemm64<<<dim3(RR / 64, D / 64), 128, 0, stream>>>(
        qln, D, Wqt, D, nullptr, qF, D, RR, D, D, 0, 0);
    k_pack_q<<<grid1(RR * D), 256, 0, stream>>>(qF, qpad, RR * D);

    // dots = q @ k^T (batched), inverted softmax over slots, renormalize
    k_gemm_skinny<<<dim3(NN / 128, 1, B), 128, 0, stream>>>(
        qpad, (size_t)16 * D, D, kbuf, (size_t)NN * D, D,
        dots, (size_t)16 * NN, NN, NN, D, 16);
    k_softmax_col<<<grid1(B * NN), 256, 0, stream>>>(dots);
    k_rowsum<<<RR, 256, 0, stream>>>(dots, sums);
    k_attn_norm<<<grid1(B * NS * NN), 256, 0, stream>>>(
        dots, sums, attnB, (l == 3) ? ((float*)d_out + (size_t)RR * D) : nullptr);

    // updates = attn @ v  (B operand is pre-transposed vt)
    k_gemm_skinny<<<dim3(D / 128, 1, B), 128, 0, stream>>>(
        attnB, (size_t)16 * NN, NN, vt, (size_t)D * NN, NN,
        updF, (size_t)NS * D, D, D, NN, NS);
    k_cvt_bf16<<<grid1(RR * D), 256, 0, stream>>>(updF, updB, RR * D);

    // GRU cell
    k_gemm64<<<dim3(RR / 64, G3 / 64), 128, 0, stream>>>(
        updB, D, Wihb, D, gbih, gi, G3, RR, G3, D, 0, 0);
    k_gemm64<<<dim3(RR / 64, G3 / 64), 128, 0, stream>>>(
        slotsB, D, Whhb, D, gbhh, gh, G3, RR, G3, D, 0, 0);
    k_gru_combine<<<grid1(RR * D), 256, 0, stream>>>(gi, gh, slotsF, newh, RR * D);
    k_ln_rows<<<RR, 256, 0, stream>>>(newh, ni_g, ni_b, slotsF, slotsB, D);

    // SwiGLU FFN + residual
    k_gemm64<<<dim3(RR / 64, HP / 64), 128, 0, stream>>>(
        slotsB, D, w1t + (size_t)l * HP * D, D, b1p + (size_t)l * HP, h1, HP, RR, HP, D, 0, 0);
    k_gemm64<<<dim3(RR / 64, HP / 64), 128, 0, stream>>>(
        slotsB, D, w2t + (size_t)l * HP * D, D, b2p + (size_t)l * HP, h2, HP, RR, HP, D, 0, 0);
    k_swiglu<<<grid1(RR * HP), 256, 0, stream>>>(h1, h2, gact, RR * HP);
    k_gemm64<<<dim3(RR / 64, D / 64), 128, 0, stream>>>(
        gact, HP, w3t + (size_t)l * D * HP, HP, f_b3 + (size_t)l * D, ffno, D, RR, D, HP, 0, 0);
    k_add<<<grid1(RR * D), 256, 0, stream>>>(slotsF, ffno, RR * D);
  }

  // outputs: slots [32,8,512] f32, then attn_map [32,4096,8] f32 (already written)
  k_copyf<<<grid1(RR * D), 256, 0, stream>>>((float*)d_out, slotsF, RR * D);
}
